// Block_38878043964083
// MI455X (gfx1250) — matrix operand order repaired and checked by reading, not compiled, not run on hardware
//
#include <hip/hip_runtime.h>
#include <cstdint>
#include <cstddef>

// ---------------------------------------------------------------------------
// Types for CDNA5 WMMA (wave32)
// ---------------------------------------------------------------------------
typedef __bf16 bf16_t;
typedef bf16_t v16bf __attribute__((ext_vector_type(16)));
typedef bf16_t v8bf  __attribute__((ext_vector_type(8)));
typedef float  v8f   __attribute__((ext_vector_type(8)));

#define DEV static __device__ __forceinline__

// Model dims
static constexpr int B_  = 8;
static constexpr int T_  = 1024;
static constexpr int C_  = 768;
static constexpr int NH_ = 12;
static constexpr int HD_ = 64;
static constexpr int M_  = B_ * T_;          // 8192 rows

// ---------------------------------------------------------------------------
// gfx1250 async global->LDS copy (ASYNCcnt path) via inline asm.
// __has_builtin is used only as a feature probe for the instruction's
// existence on this target; the asm bypasses the builtin's pointer typing.
// ---------------------------------------------------------------------------
#if defined(__has_builtin)
#if __has_builtin(__builtin_amdgcn_global_load_async_to_lds_b128)
#define USE_ASYNC_LDS 1
#endif
#endif
#ifndef USE_ASYNC_LDS
#define USE_ASYNC_LDS 0
#endif

// Copy 16 bytes global -> LDS. Async when available (tracked by ASYNCcnt).
DEV void copy16_g2l(const unsigned short* g, unsigned short* l) {
#if USE_ASYNC_LDS
    // Flat pointer to LDS: bits[31:0] are the LDS byte offset (aperture above).
    unsigned           lds_off = (unsigned)(uintptr_t)l;
    unsigned long long gaddr   = (unsigned long long)(uintptr_t)g;
    asm volatile("global_load_async_to_lds_b128 %0, %1, off"
                 :: "v"(lds_off), "v"(gaddr) : "memory");
#else
    *reinterpret_cast<uint4*>(l) = *reinterpret_cast<const uint4*>(g);
#endif
}
DEV void wait_async_le4() {
#if USE_ASYNC_LDS
    asm volatile("s_wait_asynccnt 0x4" ::: "memory");
#endif
}
DEV void wait_async_le0() {
#if USE_ASYNC_LDS
    asm volatile("s_wait_asynccnt 0x0" ::: "memory");
#endif
}

// ---------------------------------------------------------------------------
// bf16 helpers (kept as raw u16 outside WMMA operands)
// ---------------------------------------------------------------------------
DEV unsigned short f2bf(float f) {
    union { float f; unsigned u; } v; v.f = f;
    unsigned u = v.u;
    return (unsigned short)((u + 0x7FFFu + ((u >> 16) & 1u)) >> 16);
}

DEV v16bf frag_cat(v8bf lo, v8bf hi) {
    return __builtin_shufflevector(lo, hi, 0,1,2,3,4,5,6,7,8,9,10,11,12,13,14,15);
}
// A-matrix 16x32 bf16 fragment: lane holds row M=lane&15; element e -> k =
// (e&7) + 16*(e>>3) + 8*(lane>=16).  Two contiguous 8-halfword chunks.
DEV v16bf load_frag_a(const unsigned short* base, int hh) {
    v8bf lo = *reinterpret_cast<const v8bf*>(base + 8 * hh);
    v8bf hi = *reinterpret_cast<const v8bf*>(base + 16 + 8 * hh);
    return frag_cat(lo, hi);
}
// B-matrix 32x16 bf16 fragment (stored N-major, K contiguous per row):
// lane holds col N=lane&15; element e -> k = (e&7) + 16*(e>>3) + 8*(lane>=16).
DEV v16bf load_frag_b(const unsigned short* base, int hh) {
    v8bf lo = *reinterpret_cast<const v8bf*>(base + 8 * hh);
    v8bf hi = *reinterpret_cast<const v8bf*>(base + 16 + 8 * hh);
    return frag_cat(lo, hi);
}
DEV v8f wmma_bf16(v16bf a, v16bf b, v8f c) {
    return __builtin_amdgcn_wmma_f32_16x16x32_bf16(false, a, false, b,
                                                   (short)0, c, false, false);
}

// ---------------------------------------------------------------------------
// Weight cast + transpose: wT[n*K + k] = (bf16) W[k*N + n]
// (one-time pass; lets GEMM stage B tiles as contiguous async b128 copies)
// ---------------------------------------------------------------------------
__global__ __launch_bounds__(256)
void castT_f32_bf16(const float* __restrict__ in, unsigned short* __restrict__ out,
                    int K, int N) {
    int i = blockIdx.x * 256 + threadIdx.x;
    if (i < K * N) {
        int n = i / K, k = i - n * K;
        out[i] = f2bf(in[(size_t)k * N + n]);
    }
}

// ---------------------------------------------------------------------------
// LayerNorm over C=768, one row per block (256 threads x 3 elems), bf16 out
// ---------------------------------------------------------------------------
__global__ __launch_bounds__(256)
void layernorm_bf16_kernel(const float* __restrict__ x, const float* __restrict__ g,
                           const float* __restrict__ bt, unsigned short* __restrict__ out) {
    __shared__ float red[256];
    const int row = blockIdx.x;
    const int t = threadIdx.x;
    const float* xr = x + (size_t)row * C_;
    float v0 = xr[t], v1 = xr[t + 256], v2 = xr[t + 512];

    red[t] = v0 + v1 + v2;
    __syncthreads();
    #pragma unroll
    for (int s = 128; s > 0; s >>= 1) { if (t < s) red[t] += red[t + s]; __syncthreads(); }
    const float mu = red[0] * (1.0f / C_);
    __syncthreads();

    float d0 = v0 - mu, d1 = v1 - mu, d2 = v2 - mu;
    red[t] = d0 * d0 + d1 * d1 + d2 * d2;
    __syncthreads();
    #pragma unroll
    for (int s = 128; s > 0; s >>= 1) { if (t < s) red[t] += red[t + s]; __syncthreads(); }
    const float rstd = rsqrtf(red[0] * (1.0f / C_) + 1e-5f);

    unsigned short* orow = out + (size_t)row * C_;
    orow[t]       = f2bf(d0 * rstd * g[t]       + bt[t]);
    orow[t + 256] = f2bf(d1 * rstd * g[t + 256] + bt[t + 256]);
    orow[t + 512] = f2bf(d2 * rstd * g[t + 512] + bt[t + 512]);
}

// ---------------------------------------------------------------------------
// BF16 WMMA GEMM: out = act(A[MxK] @ WT[NxK]^T + bias [+ resid])
// Block tile 128x128, 256 threads = 8 waves, wave tile 32x64 (2x4 of 16x16).
// Double-buffered LDS staging via async global->LDS copies (ASYNCcnt).
// MODE 0: bf16 store   MODE 1: GELU(tanh) -> bf16   MODE 2: +resid -> f32
// ---------------------------------------------------------------------------
template <int MODE>
__global__ __launch_bounds__(256)
void gemm_bf16_wmma(const unsigned short* __restrict__ A,
                    const unsigned short* __restrict__ WT,   // N x K (transposed weight)
                    const float* __restrict__ bias,
                    const float* __restrict__ resid,
                    void* __restrict__ out,
                    int M, int N, int K) {
    __shared__ unsigned short at [2][128 * 40];  // [m][k], pitch 40 halves (80B rows)
    __shared__ unsigned short btl[2][128 * 40];  // [n][k], pitch 40

    const int t = threadIdx.x;
    const int lane = t & 31, wv = t >> 5;
    const int ln15 = lane & 15, hh = lane >> 4;
    const int wm = wv >> 1, wn = wv & 1;          // 4 waves in M, 2 in N
    const int m0 = blockIdx.y * 128, n0 = blockIdx.x * 128;

    v8f acc[2][4];
    #pragma unroll
    for (int i = 0; i < 2; ++i)
        #pragma unroll
        for (int j = 0; j < 4; ++j)
            acc[i][j] = (v8f){0.f,0.f,0.f,0.f,0.f,0.f,0.f,0.f};

    // Each thread stages one 16-halfword chunk of A and of B per K-tile.
    const int srow = t >> 1, skc = (t & 1) * 16;   // 128 rows x 32 k
    const unsigned short* aSrc = A  + (size_t)(m0 + srow) * K + skc;
    const unsigned short* bSrc = WT + (size_t)(n0 + srow) * K + skc;
    unsigned short* aDst0 = &at [0][srow * 40 + skc];
    unsigned short* bDst0 = &btl[0][srow * 40 + skc];
    unsigned short* aDst1 = &at [1][srow * 40 + skc];
    unsigned short* bDst1 = &btl[1][srow * 40 + skc];

    const int ntiles = K >> 5;

    // Prologue: stage tile 0 into buffer 0
    copy16_g2l(aSrc,     aDst0);
    copy16_g2l(aSrc + 8, aDst0 + 8);
    copy16_g2l(bSrc,     bDst0);
    copy16_g2l(bSrc + 8, bDst0 + 8);

    for (int i = 0; i < ntiles; ++i) {
        __syncthreads();   // everyone done reading buffer (i+1)&1 from tile i-1
        if (i + 1 < ntiles) {
            const int k1 = (i + 1) << 5;
            unsigned short* aD = ((i + 1) & 1) ? aDst1 : aDst0;
            unsigned short* bD = ((i + 1) & 1) ? bDst1 : bDst0;
            copy16_g2l(aSrc + k1,     aD);
            copy16_g2l(aSrc + k1 + 8, aD + 8);
            copy16_g2l(bSrc + k1,     bD);
            copy16_g2l(bSrc + k1 + 8, bD + 8);
            wait_async_le4();          // oldest 4 (tile i) have landed
        } else {
            wait_async_le0();
        }
        __syncthreads();   // all waves' tile-i copies visible

        const unsigned short* aBuf = at [i & 1];
        const unsigned short* bBuf = btl[i & 1];
        v16bf af[2], bfv[4];
        #pragma unroll
        for (int mt = 0; mt < 2; ++mt)
            af[mt] = load_frag_a(&aBuf[(wm * 32 + mt * 16 + ln15) * 40], hh);
        #pragma unroll
        for (int nt = 0; nt < 4; ++nt)
            bfv[nt] = load_frag_b(&bBuf[(wn * 64 + nt * 16 + ln15) * 40], hh);
        #pragma unroll
        for (int mt = 0; mt < 2; ++mt)
            #pragma unroll
            for (int nt = 0; nt < 4; ++nt)
                acc[mt][nt] = wmma_bf16(af[mt], bfv[nt], acc[mt][nt]);
    }

    // Epilogue: D layout -> lane col n = ln15, rows m = 8*hh + j
    #pragma unroll
    for (int nt = 0; nt < 4; ++nt) {
        const int colg = n0 + wn * 64 + nt * 16 + ln15;
        const float bc = bias[colg];
        #pragma unroll
        for (int mt = 0; mt < 2; ++mt) {
            const int rowg = m0 + wm * 32 + mt * 16 + 8 * hh;
            #pragma unroll
            for (int j = 0; j < 8; ++j) {
                const float val = acc[mt][nt][j] + bc;
                const size_t idx = (size_t)(rowg + j) * N + colg;
                if (MODE == 0) {
                    ((unsigned short*)out)[idx] = f2bf(val);
                } else if (MODE == 1) {
                    const float u = val;
                    const float gl = 0.5f * u *
                        (1.0f + tanhf(0.7978845608f * (u + 0.044715f * u * u * u)));
                    ((unsigned short*)out)[idx] = f2bf(gl);
                } else {
                    ((float*)out)[idx] = val + resid[idx];
                }
            }
        }
    }
}

// ---------------------------------------------------------------------------
// Flash-style attention. Grid = B*NH*(T/128), block = 256 (8 waves x 16 rows).
// qkv: bf16 [B*T, 3C] rows (q|k|v interleaved). o: bf16 [B*T, C].
// ---------------------------------------------------------------------------
__global__ __launch_bounds__(256)
void attn_wmma_kernel(const unsigned short* __restrict__ qkv,
                      const int* __restrict__ mask,
                      unsigned short* __restrict__ o) {
    constexpr int SP = 65;   // S pitch (floats)
    constexpr int PP = 72;   // P pitch (halves), 144B rows -> 16B aligned
    constexpr int VP = 72;   // V^T pitch (halves)
    __shared__ float          sS[128 * SP];
    __shared__ unsigned short sP[128 * PP];
    __shared__ unsigned short sVT[64 * VP];
    __shared__ float          sScale[128];

    const int t = threadIdx.x;
    const int lane = t & 31, wv = t >> 5;
    const int ln15 = lane & 15, hh = lane >> 4;

    const int gid = blockIdx.x;
    const int qt = gid & 7;
    const int h  = (gid >> 3) % NH_;
    const int b  = gid / (8 * NH_);
    const int qbase = qt * 128;
    const size_t RS = 3 * C_;                 // qkv row stride in elements

    // Q fragments (A layout), loaded once: 16 q-rows per wave, d split 0..31/32..63
    const unsigned short* qrow =
        qkv + (size_t)(b * T_ + qbase + wv * 16 + ln15) * RS + h * HD_;
    const v16bf qf0 = load_frag_a(qrow, hh);
    const v16bf qf1 = load_frag_a(qrow + 32, hh);

    v8f acc[4];
    #pragma unroll
    for (int i = 0; i < 4; ++i) acc[i] = (v8f){0.f,0.f,0.f,0.f,0.f,0.f,0.f,0.f};

    const float NEG_INF = -__builtin_inff();
    float m_reg = NEG_INF, l_reg = 0.0f;      // per-row state for tid<128

    const int vkey = t >> 2, vdc = (t & 3) * 16;   // V staging assignment
    const unsigned short* vSrc =
        qkv + (size_t)(b * T_ + vkey) * RS + 2 * C_ + h * HD_ + vdc;

    for (int kt = 0; kt < 16; ++kt) {
        const int kbase = kt * 64;

        // ---- stage V tile transposed: sVT[d][key] ----
        {
            const uint4* vs = reinterpret_cast<const uint4*>(vSrc + (size_t)kbase * RS);
            uint4 p0 = vs[0], p1 = vs[1];
            unsigned short tmp[16];
            *reinterpret_cast<uint4*>(tmp)     = p0;
            *reinterpret_cast<uint4*>(tmp + 8) = p1;
            #pragma unroll
            for (int i = 0; i < 16; ++i) sVT[(vdc + i) * VP + vkey] = tmp[i];
        }

        // ---- S = (Q K^T) * 1/sqrt(HD), masked, into LDS ----
        #pragma unroll
        for (int nt = 0; nt < 4; ++nt) {
            const int keyn = kbase + nt * 16 + ln15;
            const unsigned short* krow =
                qkv + (size_t)(b * T_ + keyn) * RS + C_ + h * HD_;
            v8f s = (v8f){0.f,0.f,0.f,0.f,0.f,0.f,0.f,0.f};
            s = wmma_bf16(qf0, load_frag_b(krow, hh), s);
            s = wmma_bf16(qf1, load_frag_b(krow + 32, hh), s);
            const int mv = mask[b * T_ + keyn];
            #pragma unroll
            for (int j = 0; j < 8; ++j) {
                const float val = (mv == 0) ? NEG_INF : s[j] * 0.125f;
                sS[(wv * 16 + 8 * hh + j) * SP + nt * 16 + ln15] = val;
            }
        }
        __syncthreads();

        // ---- online softmax: thread t<128 owns query row t ----
        if (t < 128) {
            float tmax = NEG_INF;
            for (int j = 0; j < 64; ++j) tmax = fmaxf(tmax, sS[t * SP + j]);
            const float mnew = fmaxf(m_reg, tmax);
            float lad = 0.0f, scl;
            if (mnew == NEG_INF) {
                scl = 1.0f;
                for (int j = 0; j < 64; ++j) sP[t * PP + j] = 0;
            } else {
                scl = __expf(m_reg - mnew);   // exp(-inf)=0 first time
                for (int j = 0; j < 64; ++j) {
                    const float p = __expf(sS[t * SP + j] - mnew);
                    lad += p;
                    sP[t * PP + j] = f2bf(p);
                }
            }
            l_reg = l_reg * scl + lad;
            m_reg = mnew;
            sScale[t] = scl;
        }
        __syncthreads();

        // ---- O = O*scale + P @ V_tile ----
        float rs[8];
        #pragma unroll
        for (int j = 0; j < 8; ++j) rs[j] = sScale[wv * 16 + 8 * hh + j];
        #pragma unroll
        for (int nt = 0; nt < 4; ++nt)
            #pragma unroll
            for (int j = 0; j < 8; ++j) acc[nt][j] *= rs[j];

        const unsigned short* prow = &sP[(wv * 16 + ln15) * PP];
        const v16bf pf0 = load_frag_a(prow, hh);
        const v16bf pf1 = load_frag_a(prow + 32, hh);
        #pragma unroll
        for (int nt = 0; nt < 4; ++nt) {
            const unsigned short* vrow = &sVT[(nt * 16 + ln15) * VP];
            acc[nt] = wmma_bf16(pf0, load_frag_b(vrow, hh), acc[nt]);
            acc[nt] = wmma_bf16(pf1, load_frag_b(vrow + 32, hh), acc[nt]);
        }
        __syncthreads();
    }

    // ---- normalize and store O as bf16 into [B*T, C] layout ----
    if (t < 128) sScale[t] = (l_reg > 0.0f) ? (1.0f / l_reg) : 0.0f;
    __syncthreads();
    float inv[8];
    #pragma unroll
    for (int j = 0; j < 8; ++j) inv[j] = sScale[wv * 16 + 8 * hh + j];
    #pragma unroll
    for (int nt = 0; nt < 4; ++nt)
        #pragma unroll
        for (int j = 0; j < 8; ++j) {
            const float val = acc[nt][j] * inv[j];
            const size_t idx = (size_t)(b * T_ + qbase + wv * 16 + 8 * hh + j) * C_ +
                               h * HD_ + nt * 16 + ln15;
            o[idx] = f2bf(val);
        }
}

// ---------------------------------------------------------------------------
// Host orchestration
// ---------------------------------------------------------------------------
extern "C" void kernel_launch(void* const* d_in, const int* in_sizes, int n_in,
                              void* d_out, int out_size, void* d_ws, size_t ws_size,
                              hipStream_t stream) {
    (void)in_sizes; (void)n_in; (void)out_size; (void)ws_size;

    const float* x      = (const float*)d_in[0];
    const int*   amask  = (const int*)  d_in[1];
    const float* ln1_g  = (const float*)d_in[2];
    const float* ln1_b  = (const float*)d_in[3];
    const float* W_attn = (const float*)d_in[4];
    const float* b_attn = (const float*)d_in[5];
    const float* W_proj = (const float*)d_in[6];
    const float* b_proj = (const float*)d_in[7];
    const float* ln2_g  = (const float*)d_in[8];
    const float* ln2_b  = (const float*)d_in[9];
    const float* W_fc   = (const float*)d_in[10];
    const float* b_fc   = (const float*)d_in[11];
    const float* W_fc2  = (const float*)d_in[12];
    const float* b_fc2  = (const float*)d_in[13];

    // Workspace carve-up (256B aligned)
    char* base = (char*)d_ws;
    size_t off = 0;
    auto alloc = [&](size_t bytes) -> void* {
        void* p = base + off;
        off = (off + bytes + 255) & ~(size_t)255;
        return p;
    };
    unsigned short* wAttnT = (unsigned short*)alloc((size_t)C_ * 3 * C_ * 2);  // [3C x C]
    unsigned short* wProjT = (unsigned short*)alloc((size_t)C_ * C_ * 2);      // [C x C]
    unsigned short* wFcT   = (unsigned short*)alloc((size_t)C_ * 4 * C_ * 2);  // [4C x C]
    unsigned short* wFc2T  = (unsigned short*)alloc((size_t)4 * C_ * C_ * 2);  // [C x 4C]
    unsigned short* lnbuf  = (unsigned short*)alloc((size_t)M_ * C_ * 2);      // ln1 then ln2
    unsigned short* big    = (unsigned short*)alloc((size_t)M_ * 4 * C_ * 2);  // qkv then fc1-out
    unsigned short* obuf   = (unsigned short*)alloc((size_t)M_ * C_ * 2);
    float*          x1     = (float*)        alloc((size_t)M_ * C_ * 4);

    // 1) weights -> bf16, transposed to [N x K]
    {
        int n;
        n = C_ * 3 * C_; castT_f32_bf16<<<(n + 255) / 256, 256, 0, stream>>>(W_attn, wAttnT, C_,     3 * C_);
        n = C_ * C_;     castT_f32_bf16<<<(n + 255) / 256, 256, 0, stream>>>(W_proj, wProjT, C_,     C_);
        n = C_ * 4 * C_; castT_f32_bf16<<<(n + 255) / 256, 256, 0, stream>>>(W_fc,   wFcT,   C_,     4 * C_);
        n = 4 * C_ * C_; castT_f32_bf16<<<(n + 255) / 256, 256, 0, stream>>>(W_fc2,  wFc2T,  4 * C_, C_);
    }

    // 2) LN1
    layernorm_bf16_kernel<<<M_, 256, 0, stream>>>(x, ln1_g, ln1_b, lnbuf);

    // 3) QKV = ln1 @ W_attn + b_attn  (8192 x 2304 x 768) -> bf16
    gemm_bf16_wmma<0><<<dim3(3 * C_ / 128, M_ / 128), 256, 0, stream>>>(
        lnbuf, wAttnT, b_attn, nullptr, big, M_, 3 * C_, C_);

    // 4) attention -> obuf bf16
    attn_wmma_kernel<<<B_ * NH_ * (T_ / 128), 256, 0, stream>>>(big, amask, obuf);

    // 5) x1 = x + obuf @ W_proj + b_proj  -> f32
    gemm_bf16_wmma<2><<<dim3(C_ / 128, M_ / 128), 256, 0, stream>>>(
        obuf, wProjT, b_proj, x, x1, M_, C_, C_);

    // 6) LN2
    layernorm_bf16_kernel<<<M_, 256, 0, stream>>>(x1, ln2_g, ln2_b, lnbuf);

    // 7) g = gelu(ln2 @ W_fc + b_fc)  (8192 x 3072 x 768) -> bf16 (reuse big)
    gemm_bf16_wmma<1><<<dim3(4 * C_ / 128, M_ / 128), 256, 0, stream>>>(
        lnbuf, wFcT, b_fc, nullptr, big, M_, 4 * C_, C_);

    // 8) out = x1 + g @ W_fc2 + b_fc2  (8192 x 768 x 3072) -> f32 d_out
    gemm_bf16_wmma<2><<<dim3(C_ / 128, M_ / 128), 256, 0, stream>>>(
        big, wFc2T, b_fc2, x1, d_out, M_, C_, 4 * C_);
}
